// VarLayer_11819749999269
// MI455X (gfx1250) — matrix operand order repaired and checked by reading, not compiled, not run on hardware
//
#include <hip/hip_runtime.h>
#include <hip/hip_bf16.h>
#include <math.h>

typedef __attribute__((ext_vector_type(16))) _Float16 v16h;
typedef __attribute__((ext_vector_type(8)))  _Float16 v8h;
typedef __attribute__((ext_vector_type(8)))  float    v8f;

#define PIX 576
#define HWD 24
#define NB  64      // solve batches = n*GR = 4*16
#define BW  48      // half bandwidth of ATA

// ---------------------------------------------------------------------------
// Weight prep: w (N x Kreal f32, row-major, k = ci*9+ky*3+kx) ->
// fragment-friendly f16 layout bt[kt][n][kk] (kt = k/32, kk = k%32), zero-pad.
// ---------------------------------------------------------------------------
__global__ void wprep_k(const float* __restrict__ w, _Float16* __restrict__ bt,
                        int N, int Kreal, int Kpad)
{
    int total = N * Kpad;
    for (int idx = blockIdx.x * blockDim.x + threadIdx.x; idx < total;
         idx += gridDim.x * blockDim.x) {
        int n = idx / Kpad;
        int k = idx - n * Kpad;
        float v = (k < Kreal) ? w[(size_t)n * Kreal + k] : 0.0f;
        bt[(((size_t)(k >> 5) * N + n) << 5) + (k & 31)] = (_Float16)v;
    }
}

// ---------------------------------------------------------------------------
// im2col (3x3, pad 1) -> (Nimg*576) x Kpad f16, k = ci*9 + ky*3 + kx
// srcLayout: 0 = NCHW (img,Cin,24,24); 1 = row-major MxCin (img*576+pix, Cin)
// ---------------------------------------------------------------------------
__global__ void im2col_k(const float* __restrict__ src, _Float16* __restrict__ dst,
                         int Nimg, int Cin, int Kpad, int srcLayout)
{
    int Kreal = Cin * 9;
    size_t total = (size_t)Nimg * PIX * Kpad;
    for (size_t idx = (size_t)blockIdx.x * blockDim.x + threadIdx.x; idx < total;
         idx += (size_t)gridDim.x * blockDim.x) {
        int k = (int)(idx % Kpad);
        size_t m = idx / Kpad;
        int img = (int)(m / PIX);
        int p = (int)(m - (size_t)img * PIX);
        float v = 0.0f;
        if (k < Kreal) {
            int ci = k / 9;
            int r = k - ci * 9;
            int ky = r / 3;
            int kx = r - 3 * ky;
            int py = p / HWD;
            int y = py + ky - 1;
            int x = (p - py * HWD) + kx - 1;
            if (y >= 0 && y < HWD && x >= 0 && x < HWD) {
                if (srcLayout == 0)
                    v = src[(((size_t)img * Cin + ci) * HWD + y) * HWD + x];
                else
                    v = src[((size_t)img * PIX + (y * HWD + x)) * Cin + ci];
            }
        }
        dst[idx] = (_Float16)v;
    }
}

// ---------------------------------------------------------------------------
// WMMA GEMM, NT 16-wide n-tiles per wave sharing one A fragment.
// A: MxK row-major f16 (im2col). Bt: [K/32][N][32] fragment layout.
// Per-lane fragment layouts follow CDNA5 ISA 7.12.2.
// act: 0 none, 1 leaky(0.01), 2 sigmoid.  nchw: 1 -> NCHW store, 0 -> MxN.
// ---------------------------------------------------------------------------
template <int NT>
__global__ __launch_bounds__(256) void gemm_wmma_k(
    const _Float16* __restrict__ A,
    const _Float16* __restrict__ Bt,
    const float* __restrict__ bias,
    float* __restrict__ out,
    int M, int N, int K, int act, int nchw)
{
    const int lane = threadIdx.x & 31;
    const int wid  = threadIdx.x >> 5;
    const int Mt = M >> 4;
    const int NG = N / (16 * NT);
    const int tile = blockIdx.x * 8 + wid;          // wave-uniform
    if (tile >= Mt * NG) return;                    // wave-uniform exit
    const int mt = tile / NG;
    const int ng = tile - mt * NG;
    const int hl  = lane >> 4;                      // lane half
    const int l15 = lane & 15;

    // A: lane m = l15; half 0 -> K {0..7,16..23}, half 1 -> K {8..15,24..31}
    const _Float16* arow = A + (size_t)(mt * 16 + l15) * K + hl * 8;
    // B: lane n = l15 (+ tile offset); half 0 -> K {0..7,16..23}, half 1 -> K {8..15,24..31}
    const _Float16* bbase = Bt + ((size_t)(ng * NT * 16 + l15)) * 32 + hl * 8;
    const size_t bstride = (size_t)N * 32;

    v8f c[NT];
    #pragma unroll
    for (int t = 0; t < NT; ++t) c[t] = v8f{};

    const int KT = K >> 5;
    for (int kt = 0; kt < KT; ++kt) {
        union { v16h v; v8h h[2]; } a;
        a.h[0] = *(const v8h*)(arow + kt * 32);
        a.h[1] = *(const v8h*)(arow + kt * 32 + 16);
        if (kt + 1 < KT) {
            __builtin_prefetch(arow + (kt + 1) * 32, 0, 3);
            __builtin_prefetch(bbase + (kt + 1) * bstride, 0, 3);
        }
        #pragma unroll
        for (int t = 0; t < NT; ++t) {
            union { v16h v; v8h h[2]; } b;
            const _Float16* bp = bbase + kt * bstride + (size_t)t * 16 * 32;
            b.h[0] = *(const v8h*)(bp);
            b.h[1] = *(const v8h*)(bp + 16);
            // independent accumulators -> WMMAs pipeline without hazard NOPs
            c[t] = __builtin_amdgcn_wmma_f32_16x16x32_f16(false, a.v, false, b.v,
                                                          (short)0, c[t], false, false);
        }
    }

    #pragma unroll
    for (int t = 0; t < NT; ++t) {
        int nn = ng * NT * 16 + t * 16 + l15;
        float bv = bias[nn];
        #pragma unroll
        for (int r = 0; r < 8; ++r) {
            int mm = mt * 16 + hl * 8 + r;  // C/D layout: M = r + 8*(lane/16)
            float v = c[t][r] + bv;
            if (act == 1)      v = (v > 0.0f) ? v : 0.01f * v;
            else if (act == 2) v = 1.0f / (1.0f + expf(-v));
            if (nchw) {
                int img = mm / PIX;
                int p = mm - img * PIX;
                out[((size_t)img * N + nn) * PIX + p] = v;
            } else {
                out[(size_t)mm * N + nn] = v;   // lane-contiguous (coalesced)
            }
        }
    }
}

// ---------------------------------------------------------------------------
// Squeeze-Excite: GAP -> 1x1(64->32) leaky -> 1x1(32->16) sigmoid. One block.
// ---------------------------------------------------------------------------
__global__ __launch_bounds__(256) void se_k(
    const float* __restrict__ x,
    const float* __restrict__ w1, const float* __restrict__ b1,
    const float* __restrict__ w2, const float* __restrict__ b2,
    float* __restrict__ se)
{
    __shared__ float pooled[4 * 64];
    __shared__ float hid[4 * 32];
    int tid = threadIdx.x;
    {
        int img = tid >> 6, c = tid & 63;
        const float* p = x + ((size_t)img * 64 + c) * PIX;
        float s = 0.0f;
        for (int i = 0; i < PIX; ++i) s += p[i];
        pooled[tid] = s * (1.0f / PIX);
    }
    __syncthreads();
    if (tid < 128) {
        int img = tid >> 5, cc = tid & 31;
        float s = b1[cc];
        for (int c = 0; c < 64; ++c) s += pooled[img * 64 + c] * w1[cc * 64 + c];
        hid[tid] = (s > 0.0f) ? s : 0.01f * s;
    }
    __syncthreads();
    if (tid < 64) {
        int img = tid >> 4, g = tid & 15;
        float s = b2[g];
        for (int c = 0; c < 32; ++c) s += hid[img * 32 + c] * w2[g * 32 + c];
        se[img * 16 + g] = 1.0f / (1.0f + expf(-s));
    }
}

// ---------------------------------------------------------------------------
// Build banded ATA (upper band, width 49) and ATB analytically from the
// 4-tap stencil structure (offsets 1, 24, 2, 48). Pure gather, no atomics.
// att/grad are in MxN layout: (img*576+pix, 64ch). batch b: img=b/16, g=b%16,
// tap t -> channel g*4 + t.
// ---------------------------------------------------------------------------
__device__ __forceinline__ bool tap_valid(int t, int i)
{
    if (t == 0) return ((i + 1) % HWD != 0) && (i + 1 < PIX);
    if (t == 1) return (i + 24 < PIX);
    if (t == 2) return ((i + 2) % HWD != 0) && (i + 2 < PIX);
    return (i + 48 < PIX);
}

__global__ void band_build_k(const float* __restrict__ att,
                             const float* __restrict__ grad,
                             float* __restrict__ band,
                             float* __restrict__ rhs)
{
    int idx = blockIdx.x * blockDim.x + threadIdx.x;
    if (idx >= NB * PIX) return;
    int b = idx / PIX;
    int i = idx - b * PIX;
    int img = b >> 4, g = b & 15;
    const int off[4] = {1, 24, 2, 48};

    float diag = 1e-12f, r = 0.0f;
    float od[4] = {0.0f, 0.0f, 0.0f, 0.0f};

    #pragma unroll
    for (int t = 0; t < 4; ++t) {
        int ch = g * 4 + t;
        if (tap_valid(t, i)) {                 // row (i,t): +w at i, -w at i+off
            float w = att[((size_t)img * PIX + i) * 64 + ch];
            float w2 = w * w;
            diag += w2;
            od[t] -= w2;
            r += w2 * grad[((size_t)img * PIX + i) * 64 + ch];
        }
        int j = i - off[t];                    // row (j,t) hits column i with -w
        if (j >= 0 && tap_valid(t, j)) {
            float w = att[((size_t)img * PIX + j) * 64 + ch];
            float w2 = w * w;
            diag += w2;
            r -= w2 * grad[((size_t)img * PIX + j) * 64 + ch];
        }
    }
    if (i == PIX - 1) {                        // a[-1,t,-1] = 1 for all taps
        #pragma unroll
        for (int t = 0; t < 4; ++t) {
            int ch = g * 4 + t;
            float w = att[((size_t)img * PIX + i) * 64 + ch];
            diag += w * w;
            r += w * w * grad[((size_t)img * PIX + i) * 64 + ch];
        }
    }

    float* bd = band + ((size_t)b * PIX + i) * (BW + 1);
    for (int d = 0; d <= BW; ++d) bd[d] = 0.0f;
    bd[0]  = diag;
    bd[1]  = od[0];
    bd[2]  = od[2];
    bd[24] = od[1];
    bd[48] = od[3];
    rhs[(size_t)b * PIX + i] = r;
}

// ---------------------------------------------------------------------------
// Banded LDL^T factorization + solve, one block per batch. Solution left in
// rhs, whose layout (b=img*16+g, pix) is exactly the (n,GR,h,w) flat layout.
// ---------------------------------------------------------------------------
__global__ __launch_bounds__(256) void solve_k(float* __restrict__ band,
                                               float* __restrict__ rhs)
{
    const int b = blockIdx.x;
    float* Bd = band + (size_t)b * PIX * (BW + 1);
    float* R  = rhs  + (size_t)b * PIX;
    const int tid = threadIdx.x;
    __shared__ float red[256];

    // --- factorization: right-looking banded LDL^T ---
    for (int i = 0; i < PIX; ++i) {
        int jmax = (PIX - 1 - i < BW) ? (PIX - 1 - i) : BW;
        float inv = 1.0f / Bd[i * (BW + 1)];
        // rank-1 band update: A[i+j][i+k] -= A[i][i+j]*A[i][i+k]/D
        for (int p = tid; p < (BW * (BW + 1)) / 2; p += 256) {
            int j = 1, rem = p;
            while (rem >= (BW + 1) - j) { rem -= (BW + 1) - j; ++j; }
            int k = j + rem;
            if (k <= jmax)
                Bd[(i + j) * (BW + 1) + (k - j)] -=
                    Bd[i * (BW + 1) + j] * Bd[i * (BW + 1) + k] * inv;
        }
        __syncthreads();
        if (tid >= 1 && tid <= jmax)          // store L column (scaled)
            Bd[i * (BW + 1) + tid] *= inv;
        __syncthreads();
    }

    // --- forward solve: L z = b ---
    for (int i = 0; i < PIX; ++i) {
        int jmax = (PIX - 1 - i < BW) ? (PIX - 1 - i) : BW;
        float z = R[i];
        if (tid >= 1 && tid <= jmax)
            R[i + tid] -= Bd[i * (BW + 1) + tid] * z;
        __syncthreads();
    }

    // --- backward solve: x_i = z_i/D_i - sum_j L_{i+j,i} x_{i+j} ---
    for (int i = PIX - 1; i >= 0; --i) {
        int jmax = (PIX - 1 - i < BW) ? (PIX - 1 - i) : BW;
        float part = 0.0f;
        if (tid >= 1 && tid <= jmax)
            part = Bd[i * (BW + 1) + tid] * R[i + tid];
        red[tid] = part;
        __syncthreads();
        if (tid == 0) {
            float s = 0.0f;
            for (int t = 1; t <= jmax; ++t) s += red[t];
            R[i] = R[i] / Bd[i * (BW + 1)] - s;
        }
        __syncthreads();
    }
}

// ---------------------------------------------------------------------------
// GroupNorm(1 group over GR chans + spatial) * se, one block per image.
// Input: solver layout (img*16+g, pix). Output: MxN layout (img*576+pix, 16).
// ---------------------------------------------------------------------------
__global__ __launch_bounds__(256) void gn_k(const float* __restrict__ xin,
                                            const float* __restrict__ gnw,
                                            const float* __restrict__ gnb,
                                            const float* __restrict__ se,
                                            float* __restrict__ out)
{
    const int img = blockIdx.x;
    const int tid = threadIdx.x;
    __shared__ float s1[256], s2[256];
    const float* p = xin + (size_t)img * 16 * PIX;
    float a = 0.0f, bsum = 0.0f;
    for (int i = tid; i < 16 * PIX; i += 256) {
        float v = p[i];
        a += v; bsum += v * v;
    }
    s1[tid] = a; s2[tid] = bsum;
    __syncthreads();
    for (int s = 128; s > 0; s >>= 1) {
        if (tid < s) { s1[tid] += s1[tid + s]; s2[tid] += s2[tid + s]; }
        __syncthreads();
    }
    float mu  = s1[0] * (1.0f / (16 * PIX));
    float var = s2[0] * (1.0f / (16 * PIX)) - mu * mu;
    float rstd = rsqrtf(var + 1e-5f);
    for (int i = tid; i < 16 * PIX; i += 256) {
        int g = i / PIX;
        int px = i - g * PIX;
        float v = (p[i] - mu) * rstd * gnw[g] + gnb[g];
        out[((size_t)img * PIX + px) * 16 + g] = v * se[img * 16 + g];
    }
}

// ---------------------------------------------------------------------------
// Host launcher
// ---------------------------------------------------------------------------
static inline size_t aup(size_t x) { return (x + 255) & ~size_t(255); }

extern "C" void kernel_launch(void* const* d_in, const int* in_sizes, int n_in,
                              void* d_out, int out_size, void* d_ws, size_t ws_size,
                              hipStream_t stream)
{
    (void)in_sizes; (void)n_in; (void)out_size; (void)ws_size;
    const float* x       = (const float*)d_in[0];
    // d_in[1] = stencil 'a' -- not needed: band structure is analytic
    const float* grad_w1 = (const float*)d_in[2];
    const float* grad_b1 = (const float*)d_in[3];
    const float* grad_w2 = (const float*)d_in[4];
    const float* grad_b2 = (const float*)d_in[5];
    const float* att_w1  = (const float*)d_in[6];
    const float* att_b1  = (const float*)d_in[7];
    const float* att_w2  = (const float*)d_in[8];
    const float* att_b2  = (const float*)d_in[9];
    const float* se_w1   = (const float*)d_in[10];
    const float* se_b1   = (const float*)d_in[11];
    const float* se_w2   = (const float*)d_in[12];
    const float* se_b2   = (const float*)d_in[13];
    const float* gn_w    = (const float*)d_in[14];
    const float* gn_b    = (const float*)d_in[15];
    const float* post_w  = (const float*)d_in[16];
    const float* post_b  = (const float*)d_in[17];
    float* out = (float*)d_out;

    const int M = 4 * PIX;           // 2304 pixels total
    char* ws = (char*)d_ws;
    size_t o = 0;
    auto alloc = [&](size_t bytes) { void* p = ws + o; o = aup(o + bytes); return p; };

    _Float16* aim0   = (_Float16*)alloc((size_t)M * 576 * 2);
    _Float16* aim1   = (_Float16*)alloc((size_t)M * 576 * 2);
    _Float16* aimp   = (_Float16*)alloc((size_t)M * 160 * 2);
    _Float16* btA1   = (_Float16*)alloc((size_t)576 * 64 * 2);
    _Float16* btG1   = (_Float16*)alloc((size_t)576 * 64 * 2);
    _Float16* btA2   = (_Float16*)alloc((size_t)576 * 64 * 2);
    _Float16* btG2   = (_Float16*)alloc((size_t)576 * 64 * 2);
    _Float16* btP    = (_Float16*)alloc((size_t)160 * 128 * 2);
    float*    h_att  = (float*)alloc((size_t)M * 64 * 4);      // MxN
    float*    h_grad = (float*)alloc((size_t)M * 64 * 4);      // MxN
    float*    attb   = (float*)alloc((size_t)M * 64 * 4);      // MxN
    float*    gradb  = (float*)alloc((size_t)M * 64 * 4);      // MxN
    float*    seb    = (float*)alloc((size_t)4 * 16 * 4);
    float*    band   = (float*)alloc((size_t)NB * PIX * (BW + 1) * 4);
    float*    rhs    = (float*)alloc((size_t)NB * PIX * 4);
    float*    ynorm  = (float*)alloc((size_t)M * 16 * 4);      // MxN (16 ch)

    // 1) weight prep (f16 fragment layout)
    wprep_k<<<144, 256, 0, stream>>>(att_w1,  btA1, 64,  576, 576);
    wprep_k<<<144, 256, 0, stream>>>(grad_w1, btG1, 64,  576, 576);
    wprep_k<<<144, 256, 0, stream>>>(att_w2,  btA2, 64,  576, 576);
    wprep_k<<<144, 256, 0, stream>>>(grad_w2, btG2, 64,  576, 576);
    wprep_k<<<80,  256, 0, stream>>>(post_w,  btP,  128, 144, 160);

    // 2) conv1 for both branches (shared im2col of x); wave does full N=64
    im2col_k<<<5184, 256, 0, stream>>>(x, aim0, 4, 64, 576, 0);
    gemm_wmma_k<4><<<18, 256, 0, stream>>>(aim0, btA1, att_b1,  h_att,  M, 64, 576, 1, 0);
    gemm_wmma_k<4><<<18, 256, 0, stream>>>(aim0, btG1, grad_b1, h_grad, M, 64, 576, 1, 0);

    // 3) conv2: attention (sigmoid) and gradient (linear)
    im2col_k<<<5184, 256, 0, stream>>>(h_att, aim1, 4, 64, 576, 1);
    gemm_wmma_k<4><<<18, 256, 0, stream>>>(aim1, btA2, att_b2, attb, M, 64, 576, 2, 0);
    im2col_k<<<5184, 256, 0, stream>>>(h_grad, aim1, 4, 64, 576, 1);
    gemm_wmma_k<4><<<18, 256, 0, stream>>>(aim1, btG2, grad_b2, gradb, M, 64, 576, 0, 0);

    // 4) squeeze-excite gate
    se_k<<<1, 256, 0, stream>>>(x, se_w1, se_b1, se_w2, se_b2, seb);

    // 5) analytic banded normal equations + LDL^T solve (64 systems)
    band_build_k<<<144, 256, 0, stream>>>(attb, gradb, band, rhs);
    solve_k<<<NB, 256, 0, stream>>>(band, rhs);

    // 6) GroupNorm * se  (writes MxN layout for the post-conv im2col)
    gn_k<<<4, 256, 0, stream>>>(rhs, gn_w, gn_b, seb, ynorm);

    // 7) post conv (16->128), K padded 144->160; final store is NCHW (d_out)
    im2col_k<<<1440, 256, 0, stream>>>(ynorm, aimp, 4, 16, 160, 1);
    gemm_wmma_k<4><<<36, 256, 0, stream>>>(aimp, btP, post_b, out, M, 128, 160, 0, 1);
}
